// CausalSelfAttention_82755429859922
// MI455X (gfx1250) — matrix operand order repaired and checked by reading, not compiled, not run on hardware
//
#include <hip/hip_runtime.h>
#include <hip/hip_bf16.h>

// Problem constants (from reference)
#define B_   2
#define S_   2048
#define D_   1024
#define H_   16
#define HD_  64
#define WIN_ 256

typedef __attribute__((ext_vector_type(16))) __bf16 v16bf;
typedef __attribute__((ext_vector_type(8)))  __bf16 v8bf;
typedef __attribute__((ext_vector_type(8)))  float  v8f;

#define LROW 40   // bf16 elements per LDS row (80B = 5*16B -> conflict-free b128)

static __device__ __forceinline__ v8f vzero() {
    v8f z;
#pragma unroll
    for (int i = 0; i < 8; ++i) z[i] = 0.0f;
    return z;
}

// Concatenate two contiguous 16B chunks into a 16-element bf16 WMMA operand.
static __device__ __forceinline__ v16bf load_ab(const __bf16* p0, const __bf16* p1) {
    v8bf lo = *(const v8bf*)p0;
    v8bf hi = *(const v8bf*)p1;
    v16bf r;
#pragma unroll
    for (int i = 0; i < 8; ++i) { r[i] = lo[i]; r[8 + i] = hi[i]; }
    return r;
}

static __device__ __forceinline__ v8f wmma_bf16(v16bf a, v16bf b, v8f c) {
    // (neg_a, A, neg_b, B, c_mod, C, reuse_a, reuse_b)
    return __builtin_amdgcn_wmma_f32_16x16x32_bf16(false, a, false, b, (short)0, c, false, false);
}

static __device__ __forceinline__ float redmax16(float v) {
#pragma unroll
    for (int m = 1; m < 16; m <<= 1) v = fmaxf(v, __shfl_xor(v, m, 32));
    return v;
}
static __device__ __forceinline__ float redsum16(float v) {
#pragma unroll
    for (int m = 1; m < 16; m <<= 1) v += __shfl_xor(v, m, 32);
    return v;
}

// Async-copy one 64-row x 32-col bf16 weight chunk into LDS buffer `buf`.
// 256 x 16B transfers; 128 threads x 2 issues -> 2 async instrs per wave.
static __device__ __forceinline__ void issue_wchunk(
    const __bf16* __restrict__ wstrip, __bf16* WB, int kk, int buf, int tid)
{
#pragma unroll
    for (int it = 0; it < 2; ++it) {
        int e   = tid + 128 * it;          // 0..255
        int row = e >> 2;                  // 0..63
        int seg = e & 3;                   // 4 x 16B segments per row
        uint64_t g = (uint64_t)(uintptr_t)(wstrip + (size_t)row * D_ + kk + seg * 8);
        // low 32 bits of a flat pointer into __shared__ == LDS byte offset
        uint32_t l = (uint32_t)(uintptr_t)(WB + (size_t)(buf * 64 + row) * LROW + seg * 8);
        asm volatile("global_load_async_to_lds_b128 %0, %1, off"
                     :: "v"(l), "v"(g) : "memory");
    }
}

// 16x64 output strip GEMM over K=1024 with double-buffered async weight
// staging in LDS shared by the 4 waves of the block (same weight strip).
static __device__ __forceinline__ void gemm_strip(
    const __bf16* __restrict__ arow,    // per-lane A row (row m0+col), K-major
    const __bf16* __restrict__ wstrip,  // 64 rows x 1024 of W (row-major)
    __bf16* WB, int tid, int col, int half, v8f c[4])
{
    issue_wchunk(wstrip, WB, 0,  0, tid);
    issue_wchunk(wstrip, WB, 32, 1, tid);
    for (int kk = 0; kk < D_; kk += 32) {
        const int buf = (kk >> 5) & 1;
        // in-order ASYNCcnt: <=2 outstanding leaves only the next chunk in flight
        if (kk + 32 < D_) asm volatile("s_wait_asynccnt 0x2" ::: "memory");
        else              asm volatile("s_wait_asynccnt 0x0" ::: "memory");
        __syncthreads();
        __builtin_prefetch(arow + kk + 64, 0, 1);   // global_prefetch_b8 for A
        v16bf a = load_ab(arow + kk + half * 8, arow + kk + 16 + half * 8);
#pragma unroll
        for (int t = 0; t < 4; ++t) {
            const __bf16* r = WB + (size_t)(buf * 64 + t * 16 + col) * LROW + half * 8;
            v16bf b = load_ab(r, r + 16);
            c[t] = wmma_bf16(a, b, c[t]);
        }
        __syncthreads();                 // all reads of `buf` done before reuse
        if (kk + 64 < D_) issue_wchunk(wstrip, WB, kk + 64, buf, tid);
    }
}

// ---------------------------------------------------------------- fp32 -> bf16
__global__ void k_cvt_bf16(const float* __restrict__ s, __bf16* __restrict__ d, int n) {
    int i = (blockIdx.x * blockDim.x + threadIdx.x) * 8;
    if (i + 8 > n) return;
    const float4* p = (const float4*)(s + i);
    float4 f0 = p[0], f1 = p[1];
    v8bf o;
    o[0] = (__bf16)f0.x; o[1] = (__bf16)f0.y; o[2] = (__bf16)f0.z; o[3] = (__bf16)f0.w;
    o[4] = (__bf16)f1.x; o[5] = (__bf16)f1.y; o[6] = (__bf16)f1.z; o[7] = (__bf16)f1.w;
    *(v8bf*)(d + i) = o;
}

// ------------------------------------------------- fused QKV projection + RoPE
// Block: 4 waves, consecutive 16-row tiles, same (head, q|k|v) -> shared weights.
__global__ __launch_bounds__(128) void k_qkv_rope(
    const __bf16* __restrict__ xb,
    const __bf16* __restrict__ wq, const __bf16* __restrict__ wk,
    const __bf16* __restrict__ wv,
    __bf16* __restrict__ q, __bf16* __restrict__ kout, __bf16* __restrict__ vT)
{
    __shared__ __bf16 WB[2 * 64 * LROW];            // 10 KB double buffer
    const int tid  = threadIdx.x;
    const int lane = tid & 31;
    const int col  = lane & 15;
    const int half = lane >> 4;
    const int w    = blockIdx.x * 4 + (tid >> 5);
    const int mt   = w & 255;          // 256 row tiles of 16 over B*S=4096
    const int h    = (w >> 8) & 15;    // head       (uniform per block)
    const int sel  = w >> 12;          // 0=q 1=k 2=v (uniform per block)
    const int m0   = mt * 16;

    const __bf16* wmat = (sel == 0) ? wq : ((sel == 1) ? wk : wv);
    const __bf16* arow = xb + (size_t)(m0 + col) * D_;

    v8f c[4];
#pragma unroll
    for (int t = 0; t < 4; ++t) c[t] = vzero();

    gemm_strip(arow, wmat + (size_t)h * HD_ * D_, WB, tid, col, half, c);

    const int b      = m0 >> 11;        // S_=2048; tiles never straddle batch
    const int srow0  = m0 & (S_ - 1);
    const size_t bh  = (size_t)b * H_ + h;

    if (sel < 2) {
        __bf16* outp = (sel == 0) ? q : kout;
#pragma unroll
        for (int t = 0; t < 4; ++t) {
            const int d  = t * 16 + col;        // 0..63 within head
            const int fi = d & 31;
            // inv_freq = 10000^(-2*fi/64) = exp(-fi * ln(10000)/32)
            const float inv = __expf(-(float)fi * 0.2878231366242558f);
#pragma unroll
            for (int r = 0; r < 8; ++r) {
                const int M = r + 8 * half;
                const int s = srow0 + M;
                float cs, sn;
                __sincosf((float)s * inv, &sn, &cs);
                float val = c[t][r];
                float partner = __shfl_xor(val, 1, 32);       // element at d^1
                float rot = (d & 1) ? partner : -partner;      // rotate_half
                float o = val * cs + rot * sn;
                outp[(bh * S_ + s) * HD_ + d] = (__bf16)o;
            }
        }
    } else {
        // v stored transposed [B][H][HD][S] for contiguous P@V B-loads
#pragma unroll
        for (int t = 0; t < 4; ++t) {
            const int d = t * 16 + col;
#pragma unroll
            for (int r = 0; r < 8; ++r) {
                const int M = r + 8 * half;
                const int s = srow0 + M;
                vT[(bh * HD_ + d) * S_ + s] = (__bf16)c[t][r];
            }
        }
    }
}

// --------------------------------------- sliding-window flash attention (fwd)
// One wave per (b, h, 16-query tile); online softmax over key blocks of 32.
__global__ __launch_bounds__(128) void k_attn(
    const __bf16* __restrict__ q, const __bf16* __restrict__ kmat,
    const __bf16* __restrict__ vT, __bf16* __restrict__ y)
{
    __shared__ __bf16 Pb[4][16][32];    // per-wave P-tile relayout buffer (4 KB)
    const int lane = threadIdx.x & 31;
    const int col  = lane & 15;
    const int half = lane >> 4;
    const int wv_  = threadIdx.x >> 5;
    const int w    = blockIdx.x * 4 + wv_;
    const int mt   = w & 127;
    const int h    = (w >> 7) & 15;
    const int b    = w >> 11;
    const int s0   = mt * 16;
    const size_t bh = (size_t)b * H_ + h;

    const __bf16* qb = q    + bh * (size_t)S_ * HD_;
    const __bf16* kb = kmat + bh * (size_t)S_ * HD_;
    const __bf16* vb = vT   + bh * (size_t)HD_ * S_;

    // Q A-tiles (K = head dim 64 -> two 16x32 operands), reused every block
    const __bf16* qrow = qb + (size_t)(s0 + col) * HD_;
    v16bf a0 = load_ab(qrow + half * 8,      qrow + 16 + half * 8);
    v16bf a1 = load_ab(qrow + 32 + half * 8, qrow + 48 + half * 8);

    float mrun[8], lrun[8];
    v8f acc[4];
#pragma unroll
    for (int r = 0; r < 8; ++r) { mrun[r] = -1e30f; lrun[r] = 0.0f; }
#pragma unroll
    for (int t = 0; t < 4; ++t) acc[t] = vzero();

    int kstart = s0 - (WIN_ - 1);
    if (kstart < 0) kstart = 0;
    kstart &= ~31;
    const int kend = s0 + 15;

    for (int k0 = kstart; k0 <= kend; k0 += 32) {
        // ---- scores: two 16x16 tiles, K=64 split in 2x32
        const __bf16* kr0 = kb + (size_t)(k0 + col) * HD_;
        const __bf16* kr1 = kb + (size_t)(k0 + 16 + col) * HD_;
        v16bf b00 = load_ab(kr0 + half * 8,      kr0 + 16 + half * 8);
        v16bf b01 = load_ab(kr0 + 32 + half * 8, kr0 + 48 + half * 8);
        v16bf b10 = load_ab(kr1 + half * 8,      kr1 + 16 + half * 8);
        v16bf b11 = load_ab(kr1 + 32 + half * 8, kr1 + 48 + half * 8);
        v8f sc0 = vzero(), sc1 = vzero();
        sc0 = wmma_bf16(a0, b00, sc0);
        sc0 = wmma_bf16(a1, b01, sc0);
        sc1 = wmma_bf16(a0, b10, sc1);
        sc1 = wmma_bf16(a1, b11, sc1);

        // ---- online softmax update + write P tile (bf16) to LDS
#pragma unroll
        for (int r = 0; r < 8; ++r) {
            const int M    = r + 8 * half;
            const int qpos = s0 + M;
            const int key0 = k0 + col, key1 = key0 + 16;
            float x0 = sc0[r] * 0.125f;              // 1/sqrt(64)
            float x1 = sc1[r] * 0.125f;
            bool al0 = (key0 <= qpos) && (qpos - key0 < WIN_);
            bool al1 = (key1 <= qpos) && (qpos - key1 < WIN_);
            float tm = fmaxf(al0 ? x0 : -1e30f, al1 ? x1 : -1e30f);
            tm = redmax16(tm);
            float nm   = fmaxf(mrun[r], tm);
            float corr = __expf(mrun[r] - nm);
            mrun[r] = nm;
            float p0 = al0 ? __expf(x0 - nm) : 0.0f;
            float p1 = al1 ? __expf(x1 - nm) : 0.0f;
            lrun[r] = lrun[r] * corr + redsum16(p0 + p1);
#pragma unroll
            for (int t = 0; t < 4; ++t) acc[t][r] *= corr;
            Pb[wv_][M][col]      = (__bf16)p0;
            Pb[wv_][M][col + 16] = (__bf16)p1;
        }
        asm volatile("s_wait_dscnt 0x0" ::: "memory");   // cross-lane LDS RAW
        __builtin_amdgcn_wave_barrier();

        // ---- re-load P in A-layout (16x32, K = key index)
        const __bf16* prow = &Pb[wv_][col][0];
        v16bf pa = load_ab(prow + half * 8, prow + 16 + half * 8);
        asm volatile("" ::: "memory");                   // keep loads before next stores

        // ---- O += P @ V  (4 column tiles of HD)
#pragma unroll
        for (int t = 0; t < 4; ++t) {
            const __bf16* vrow = vb + (size_t)(t * 16 + col) * S_ + k0 + half * 8;
            v16bf bv = load_ab(vrow, vrow + 16);
            acc[t] = wmma_bf16(pa, bv, acc[t]);
        }
    }

    // ---- normalize and store y as bf16 [B,S,H*HD]
#pragma unroll
    for (int t = 0; t < 4; ++t) {
#pragma unroll
        for (int r = 0; r < 8; ++r) {
            const int M = r + 8 * half;
            float o = acc[t][r] / lrun[r];
            y[((size_t)b * S_ + s0 + M) * D_ + h * HD_ + t * 16 + col] = (__bf16)o;
        }
    }
}

// ------------------------------------------------------ output projection GEMM
__global__ __launch_bounds__(128) void k_oproj(
    const __bf16* __restrict__ yb, const __bf16* __restrict__ wo,
    float* __restrict__ out)
{
    __shared__ __bf16 WB[2 * 64 * LROW];
    const int tid  = threadIdx.x;
    const int lane = tid & 31;
    const int col  = lane & 15;
    const int half = lane >> 4;
    const int w    = blockIdx.x * 4 + (tid >> 5);
    const int mt   = w & 255;       // 256 row tiles
    const int nt   = w >> 8;        // 16 column strips of 64 (uniform per block)
    const int m0   = mt * 16;

    const __bf16* arow = yb + (size_t)(m0 + col) * D_;

    v8f c[4];
#pragma unroll
    for (int t = 0; t < 4; ++t) c[t] = vzero();

    gemm_strip(arow, wo + (size_t)nt * 64 * D_, WB, tid, col, half, c);

#pragma unroll
    for (int t = 0; t < 4; ++t)
#pragma unroll
        for (int r = 0; r < 8; ++r)
            out[(size_t)(m0 + r + 8 * half) * D_ + nt * 64 + t * 16 + col] = c[t][r];
}

// ---------------------------------------------------------------------- driver
extern "C" void kernel_launch(void* const* d_in, const int* in_sizes, int n_in,
                              void* d_out, int out_size, void* d_ws, size_t ws_size,
                              hipStream_t stream) {
    const float* x  = (const float*)d_in[0];
    const float* Wq = (const float*)d_in[1];
    const float* Wk = (const float*)d_in[2];
    const float* Wv = (const float*)d_in[3];
    const float* Wo = (const float*)d_in[4];

    char* ws = (char*)d_ws;
    __bf16* xb  = (__bf16*)(ws);                    //  8 MB: [4096][1024]
    __bf16* wqb = (__bf16*)(ws + 8388608);          //  2 MB
    __bf16* wkb = (__bf16*)(ws + 10485760);         //  2 MB
    __bf16* wvb = (__bf16*)(ws + 12582912);         //  2 MB
    __bf16* wob = (__bf16*)(ws + 14680064);         //  2 MB
    __bf16* qb  = (__bf16*)(ws + 16777216);         //  8 MB: [B][H][S][HD]
    __bf16* kbf = (__bf16*)(ws + 25165824);         //  8 MB: [B][H][S][HD]
    __bf16* vTb = (__bf16*)(ws + 33554432);         //  8 MB: [B][H][HD][S]
    __bf16* yb  = (__bf16*)(ws + 41943040);         //  8 MB: [4096][1024]

    const int NX = B_ * S_ * D_;                    // 4194304
    const int NW = D_ * D_;                         // 1048576

    k_cvt_bf16<<<NX / 8 / 256, 256, 0, stream>>>(x,  xb,  NX);
    k_cvt_bf16<<<NW / 8 / 256, 256, 0, stream>>>(Wq, wqb, NW);
    k_cvt_bf16<<<NW / 8 / 256, 256, 0, stream>>>(Wk, wkb, NW);
    k_cvt_bf16<<<NW / 8 / 256, 256, 0, stream>>>(Wv, wvb, NW);
    k_cvt_bf16<<<NW / 8 / 256, 256, 0, stream>>>(Wo, wob, NW);

    // 3 matrices x 16 heads x 256 m-tiles = 12288 waves, 4 waves/block
    k_qkv_rope<<<3072, 128, 0, stream>>>(xb, wqb, wkb, wvb, qb, kbf, vTb);

    // 2 x 16 x 128 = 4096 waves
    k_attn<<<1024, 128, 0, stream>>>(qb, kbf, vTb, yb);

    // 256 m-tiles x 16 n-strips = 4096 waves
    k_oproj<<<1024, 128, 0, stream>>>(yb, wob, (float*)d_out);
}
